// Attention_5566277615942
// MI455X (gfx1250) — matrix operand order repaired and checked by reading, not compiled, not run on hardware
//
#include <hip/hip_runtime.h>
#include <hip/hip_bf16.h>
#include <math.h>
#include <stdint.h>

// ---------------------------------------------------------------------------
// Enformer-style relative-position attention for MI455X (gfx1250, wave32).
// All GEMMs run on v_wmma_f32_16x16x32_f16, with B tiles staged into LDS via
// the CDNA5 async global->LDS DMA path (global_load_async_to_lds_b64 +
// s_wait_asynccnt, double-buffered). The attention core is fused per
// (batch*head, 16-row tile): rel-band WMMA -> content WMMA + shift-gather ->
// block softmax -> PV WMMA, never materializing (b,h,n,2n-1) in HBM.
// ---------------------------------------------------------------------------

typedef __attribute__((ext_vector_type(16))) _Float16 v16h;
typedef __attribute__((ext_vector_type(8)))  _Float16 v8h;
typedef __attribute__((ext_vector_type(8)))  float    v8f;

#define BATCH 2
#define SEQ   1536
#define DIM   1536
#define HEADS 8
#define DK    64
#define DV    192
#define HDK   512     // HEADS*DK
#define HDV   1536    // HEADS*DV
#define NRPF  192     // rel pos features
#define NB    32      // num_basis = NRPF/6
#define LREL  3071    // 2n-1
#define LPAD  3072
#define BH    (BATCH*HEADS)
#define JT    (SEQ/16)        // 96 column tiles
#define BANDT 97              // ceil((n+15)/16)
#define BANDP (BANDT*16)      // 1552

// ---------------- WMMA helpers --------------------------------------------

__device__ __forceinline__ v8f wmma_f16(v16h a, v16h b, v8f c) {
  // D = A(16x32 f16) * B(32x16 f16) + C(16x16 f32)
  return __builtin_amdgcn_wmma_f32_16x16x32_f16(
      /*neg_a=*/false, a, /*neg_b=*/false, b,
      /*c_mod=*/(short)0, c, /*reuse_a=*/false, /*reuse_b=*/false);
}

// A fragment, row-major A[M][lda]. 16-bit A layout: lane L holds row M=L%16;
// lanes 0-15: K = {k0..k0+7, k0+16..k0+23}; lanes 16-31: +8.
__device__ __forceinline__ v16h frag_a(const _Float16* A, int lda, int mBase,
                                       int k0, int lane, int mMax) {
  int row = mBase + (lane & 15);
  row = row > mMax ? mMax : row;
  const _Float16* p = A + (size_t)row * lda + (k0 + ((lane >> 4) << 3));
  v8h lo = *(const v8h*)p;
  v8h hi = *(const v8h*)(p + 16);
  v16h r;
#pragma unroll
  for (int i = 0; i < 8; ++i) { r[i] = lo[i]; r[i + 8] = hi[i]; }
  return r;
}

// B fragment from K-major storage BT[K][ldb]: lane L holds column N=L%16;
// lanes 0-15: K = {k0..k0+7, k0+16..k0+23}; lanes 16-31: +8.
__device__ __forceinline__ v16h frag_b(const _Float16* BT, int ldb,
                                       int k0, int nBase, int lane) {
  const _Float16* p = BT + (size_t)(k0 + ((lane >> 4) << 3)) * ldb + nBase + (lane & 15);
  v16h r;
#pragma unroll
  for (int i = 0; i < 8; ++i) { r[i] = p[(size_t)i * ldb]; r[i + 8] = p[(size_t)(i + 16) * ldb]; }
  return r;
}

// B fragment from an LDS-staged 32x16 tile (K-major).
__device__ __forceinline__ v16h frag_b_lds(const _Float16* sBbuf, int lane) {
  const _Float16* p = sBbuf + ((lane >> 4) << 3) * 16 + (lane & 15);
  v16h r;
#pragma unroll
  for (int i = 0; i < 8; ++i) { r[i] = p[i * 16]; r[i + 8] = p[(i + 16) * 16]; }
  return r;
}

// ---------------- precision / layout prep ---------------------------------

__global__ void cast_kernel(const float* __restrict__ in,
                            _Float16* __restrict__ out, int n) {
  int i = blockIdx.x * blockDim.x + threadIdx.x;
  if (i < n) out[i] = (_Float16)in[i];
}

// in[R][C] f32 -> out[C][R] f16 (K-major for WMMA B operands)
__global__ void transpose_cast_kernel(const float* __restrict__ in,
                                      _Float16* __restrict__ out, int R, int C) {
  int i = blockIdx.x * blockDim.x + threadIdx.x;
  if (i >= R * C) return;
  int r = i / C, c = i - r * C;
  out[(size_t)c * R + r] = (_Float16)in[i];
}

// ---------------- positional embedding (VALU + LDS reduce) ----------------

__global__ void pos_embed_kernel(_Float16* __restrict__ pos) {
  __shared__ float sG[NB];
  const int i = blockIdx.x;            // 0..LREL-1
  const int tid = threadIdx.x;         // 0..95
  const int which = tid >> 5;          // 0=exp, 1=central-mask, 2=gamma
  const int bi = tid & 31;
  const float dist = (float)(i - (SEQ - 1));
  const float adist = fabsf(dist);
  const float sgn = dist > 0.f ? 1.f : (dist < 0.f ? -1.f : 0.f);
  float feat = 0.f, prob = 0.f;
  if (which == 0) {
    const float maxr = log2f((float)SEQ);
    float hl = exp2f(3.0f + (maxr - 3.0f) * (float)bi / (float)(NB - 1));
    feat = exp2f(-adist / hl);                        // exp(-ln2/hl * d)
  } else if (which == 1) {
    float cw = exp2f((float)(bi + 1)) - 1.0f;
    feat = cw > adist ? 1.0f : 0.0f;
  } else {
    const float stddev = (float)SEQ / (2.0f * NB);    // 24
    const float startm = (float)SEQ / (float)NB;      // 48
    float mean = startm + ((float)SEQ - startm) * (float)bi / (float)(NB - 1);
    float conc = (mean / stddev) * (mean / stddev);
    float rate = mean / (stddev * stddev);
    float lu = (adist > 0.f ? (conc - 1.0f) * logf(adist) : 0.0f) - rate * adist;
    float ln = lgammaf(conc) - conc * logf(rate);
    prob = expf(lu - ln) + 1e-8f;
    sG[bi] = prob;
  }
  __syncthreads();
  if (which == 2) {
    float mx = sG[0];
#pragma unroll
    for (int t = 1; t < NB; ++t) mx = fmaxf(mx, sG[t]);
    feat = prob / mx;
  }
  pos[(size_t)i * NRPF + tid]      = (_Float16)feat;
  pos[(size_t)i * NRPF + tid + 96] = (_Float16)(sgn * feat);
}

// ---------------- generic WMMA GEMM with fused epilogues ------------------
// C(M,16-col tile) = A(M,K) * BT(K,N already K-major). 4 waves = 64 M rows
// sharing one B tile per k-step; B is staged global->LDS with the async DMA
// path (double-buffered) so the next tile's copy overlaps the current WMMA.

enum { GEMM_RELK = 0, GEMM_Q, GEMM_K, GEMM_V, GEMM_OUT };

template <int MODE>
__global__ void __launch_bounds__(128)
gemm_wmma(const _Float16* __restrict__ A, int lda,
          const _Float16* __restrict__ BT, int ldb,
          int M, int K,
          _Float16* __restrict__ o0, _Float16* __restrict__ o1,
          float* __restrict__ f0,
          const float* __restrict__ bias0, const float* __restrict__ bias1,
          float scale) {
  __shared__ __align__(32) _Float16 sB[2][32][16];   // double-buffered B tile
  const int tid   = threadIdx.x;
  const int lane  = tid & 31;
  const int wave  = tid >> 5;
  const int nBase = blockIdx.x * 16;
  const int mBase = blockIdx.y * 64 + wave * 16;
  // No early-out: all 128 threads participate in barriers; OOB rows are
  // handled by clamped loads + guarded stores (mBase < M for all launches).

  // 128 threads x 8B = one 1KB 32x16 f16 tile per issue (ASYNCcnt-tracked).
  auto issue_b = [&](int buf, int k0) {
    uint32_t laddr = (uint32_t)(size_t)(&sB[buf][0][0]) + (uint32_t)tid * 8u;
    uint64_t ga = (uint64_t)(size_t)(BT + (size_t)(k0 + (tid >> 2)) * ldb +
                                     nBase + ((tid & 3) << 2));
    asm volatile("global_load_async_to_lds_b64 %0, %1, off"
                 :: "v"(laddr), "v"(ga) : "memory");
  };

  issue_b(0, 0);
  v8f acc = {};
  const int nsteps = K >> 5;
  for (int s = 0; s < nsteps; ++s) {
    const int cur = s & 1;
    if (s + 1 < nsteps) {
      issue_b(cur ^ 1, (s + 1) << 5);
      asm volatile("s_wait_asynccnt 0x1" ::: "memory");  // current tile landed
    } else {
      asm volatile("s_wait_asynccnt 0x0" ::: "memory");
    }
    __syncthreads();
    v16h a = frag_a(A, lda, mBase, s << 5, lane, M - 1);
    v16h b = frag_b_lds(&sB[cur][0][0], lane);
    acc = wmma_f16(a, b, acc);
    __syncthreads();   // all waves done with sB[cur] before it is re-filled
  }

  const int col   = nBase + (lane & 15);
  const int rbase = mBase + ((lane >> 4) << 3);
#pragma unroll
  for (int r = 0; r < 8; ++r) {
    int row = rbase + r;
    if (row >= M) continue;
    float v = acc[r];
    if (MODE == GEMM_RELK) {                 // relkT[f][t], t-major for band B
      o0[(size_t)col * LPAD + row] = (_Float16)v;
    } else if (MODE == GEMM_Q) {             // two biased copies, head-split
      int bb = row / SEQ, j = row - bb * SEQ;
      int h = col >> 6, d = col & 63;
      size_t o = ((size_t)(bb * HEADS + h) * SEQ + j) * DK + d;
      float q = v * scale;
      o0[o] = (_Float16)(q + bias0[col]);    // q + rel_content_bias
      o1[o] = (_Float16)(q + bias1[col]);    // q + rel_pos_bias
    } else if (MODE == GEMM_K) {             // kT[bh][d][j] (K-major)
      int bb = row / SEQ, j = row - bb * SEQ;
      int h = col >> 6, d = col & 63;
      o0[((size_t)(bb * HEADS + h) * DK + d) * SEQ + j] = (_Float16)v;
    } else if (MODE == GEMM_V) {             // v[bh][j][dv] row-major
      int bb = row / SEQ, j = row - bb * SEQ;
      int h = col / DV, dv = col - h * DV;
      o0[(((size_t)(bb * HEADS + h)) * SEQ + j) * DV + dv] = (_Float16)v;
    } else {                                  // GEMM_OUT: f32 + bias
      f0[(size_t)row * DIM + col] = v + bias0[col];
    }
  }
}

// ---------------- fused relative attention --------------------------------
// One workgroup (4 waves) per (bh, 16 query rows). LDS ~100KB (<320KB/WGP).

__global__ void __launch_bounds__(128)
attn_kernel(const _Float16* __restrict__ qc,    // [BH][SEQ][DK]
            const _Float16* __restrict__ qp,    // [BH][SEQ][DK]
            const _Float16* __restrict__ khT,   // [BH][DK][SEQ]
            const _Float16* __restrict__ vh,    // [BH][SEQ][DV]
            const _Float16* __restrict__ relkT, // [HDK][LPAD]
            _Float16* __restrict__ outh)        // [B][SEQ][HDV]
{
  __shared__ _Float16 sS[16][SEQ];    // scores -> P (f16)
  __shared__ _Float16 sR[16][BANDP];  // shifted-rel band
  __shared__ float sRed[16][8];
  __shared__ float sMax[16];
  __shared__ float sInv[16];

  const int lane  = threadIdx.x & 31;
  const int wave  = threadIdx.x >> 5;
  const int iBase = blockIdx.x * 16;
  const int bh    = blockIdx.y;
  const int b = bh >> 3, h = bh & 7;
  const int tStart = SEQ - 16 - iBase;   // first rel row of band, mult of 16

  const _Float16* qcA = qc + ((size_t)bh * SEQ + iBase) * DK;
  const _Float16* qpA = qp + ((size_t)bh * SEQ + iBase) * DK;
  const _Float16* kB  = khT + (size_t)bh * DK * SEQ;
  const _Float16* rB  = relkT + (size_t)h * DK * LPAD;
  const _Float16* vB  = vh + (size_t)bh * SEQ * DV;

  // Phase 1: rel band R[ri][t-tStart] = qp_i . relk[t]  (WMMA)
  for (int tt = wave; tt < BANDT; tt += 4) {
    v8f acc = {};
#pragma unroll
    for (int k0 = 0; k0 < DK; k0 += 32) {
      v16h a = frag_a(qpA, DK, 0, k0, lane, 15);
      v16h bf = frag_b(rB, LPAD, k0, tStart + tt * 16, lane);
      acc = wmma_f16(a, bf, acc);
    }
    int cj = lane & 15, rb = (lane >> 4) << 3;
#pragma unroll
    for (int r = 0; r < 8; ++r)
      sR[rb + r][tt * 16 + cj] = (_Float16)acc[r];
  }
  __syncthreads();

  // Phase 2: S[i][j] = qc_i . k_j + R[i][j - i + 15]  (WMMA + gather)
  for (int jt = wave; jt < JT; jt += 4) {
    v8f acc = {};
#pragma unroll
    for (int k0 = 0; k0 < DK; k0 += 32) {
      v16h a = frag_a(qcA, DK, 0, k0, lane, 15);
      v16h bf = frag_b(kB, SEQ, k0, jt * 16, lane);
      acc = wmma_f16(a, bf, acc);
    }
    int cj = lane & 15, rb = (lane >> 4) << 3;
#pragma unroll
    for (int r = 0; r < 8; ++r) {
      int ri = rb + r;
      int idx = jt * 16 + cj - ri + 15;          // 0..1550
      sS[ri][jt * 16 + cj] = (_Float16)(acc[r] + (float)sR[ri][idx]);
    }
  }
  __syncthreads();

  // Phase 3: block softmax over full rows (unnormalized P, scale at end)
  {
    int r = threadIdx.x & 15, g = threadIdx.x >> 4;
    float m = -3.0e38f;
    for (int c = g; c < SEQ; c += 8) m = fmaxf(m, (float)sS[r][c]);
    sRed[r][g] = m;
    __syncthreads();
    if (threadIdx.x < 16) {
      float mm = sRed[threadIdx.x][0];
      for (int t = 1; t < 8; ++t) mm = fmaxf(mm, sRed[threadIdx.x][t]);
      sMax[threadIdx.x] = mm;
    }
    __syncthreads();
    float mm = sMax[r], s = 0.f;
    for (int c = g; c < SEQ; c += 8) {
      float e = expf((float)sS[r][c] - mm);
      sS[r][c] = (_Float16)e;
      s += e;
    }
    sRed[r][g] = s;
    __syncthreads();
    if (threadIdx.x < 16) {
      float ss = 0.f;
      for (int t = 0; t < 8; ++t) ss += sRed[threadIdx.x][t];
      sInv[threadIdx.x] = 1.0f / ss;
    }
    __syncthreads();
  }

  // Phase 4: O = P @ V  (WMMA, A fragments straight from LDS), 3 dv tiles/wave
  const _Float16* pA = &sS[0][0];
#pragma unroll
  for (int s = 0; s < 3; ++s) {
    int dvBase = (wave * 3 + s) * 16;
    v8f acc = {};
    for (int j0 = 0; j0 < SEQ; j0 += 32) {
      v16h a = frag_a(pA, SEQ, 0, j0, lane, 15);
      v16h bf = frag_b(vB, DV, j0, dvBase, lane);
      acc = wmma_f16(a, bf, acc);
    }
    int cj = lane & 15, rb = (lane >> 4) << 3;
#pragma unroll
    for (int r = 0; r < 8; ++r) {
      int ri = rb + r;
      float v = acc[r] * sInv[ri];
      outh[((size_t)b * SEQ + iBase + ri) * HDV + h * DV + dvBase + cj] = (_Float16)v;
    }
  }
}

// ---------------- host-side orchestration ---------------------------------

extern "C" void kernel_launch(void* const* d_in, const int* in_sizes, int n_in,
                              void* d_out, int out_size, void* d_ws, size_t ws_size,
                              hipStream_t stream) {
  (void)in_sizes; (void)n_in; (void)out_size; (void)ws_size;
  const float* x    = (const float*)d_in[0];
  const float* Wq   = (const float*)d_in[1];
  const float* Wk   = (const float*)d_in[2];
  const float* Wv   = (const float*)d_in[3];
  const float* Wrel = (const float*)d_in[4];
  const float* Wout = (const float*)d_in[5];
  const float* bout = (const float*)d_in[6];
  const float* rcb  = (const float*)d_in[7];
  const float* rpb  = (const float*)d_in[8];
  float* out = (float*)d_out;

  char* ws = (char*)d_ws;
  size_t off = 0;
  auto alloc = [&](size_t bytes) {
    char* p = ws + off;
    off = (off + bytes + 255) & ~(size_t)255;
    return p;
  };
  _Float16* xh    = (_Float16*)alloc((size_t)BATCH * SEQ * DIM * 2);
  _Float16* WqT   = (_Float16*)alloc((size_t)DIM * HDK * 2);
  _Float16* WkT   = (_Float16*)alloc((size_t)DIM * HDK * 2);
  _Float16* WvT   = (_Float16*)alloc((size_t)DIM * HDV * 2);
  _Float16* WoT   = (_Float16*)alloc((size_t)HDV * DIM * 2);
  _Float16* WrT   = (_Float16*)alloc((size_t)NRPF * HDK * 2);
  _Float16* pos   = (_Float16*)alloc((size_t)LREL * NRPF * 2);
  _Float16* relkT = (_Float16*)alloc((size_t)HDK * LPAD * 2);
  _Float16* qcb   = (_Float16*)alloc((size_t)BH * SEQ * DK * 2);
  _Float16* qpb   = (_Float16*)alloc((size_t)BH * SEQ * DK * 2);
  _Float16* khT   = (_Float16*)alloc((size_t)BH * DK * SEQ * 2);
  _Float16* vhb   = (_Float16*)alloc((size_t)BH * SEQ * DV * 2);
  _Float16* outh  = (_Float16*)alloc((size_t)BATCH * SEQ * HDV * 2);

  const int T = 256;
  const int M = BATCH * SEQ;           // 3072
  // precision / layout prep
  cast_kernel<<<(M * DIM + T - 1) / T, T, 0, stream>>>(x, xh, M * DIM);
  transpose_cast_kernel<<<(HDK * DIM + T - 1) / T, T, 0, stream>>>(Wq, WqT, HDK, DIM);
  transpose_cast_kernel<<<(HDK * DIM + T - 1) / T, T, 0, stream>>>(Wk, WkT, HDK, DIM);
  transpose_cast_kernel<<<(HDV * DIM + T - 1) / T, T, 0, stream>>>(Wv, WvT, HDV, DIM);
  transpose_cast_kernel<<<(DIM * HDV + T - 1) / T, T, 0, stream>>>(Wout, WoT, DIM, HDV);
  transpose_cast_kernel<<<(HDK * NRPF + T - 1) / T, T, 0, stream>>>(Wrel, WrT, HDK, NRPF);
  pos_embed_kernel<<<LREL, 96, 0, stream>>>(pos);
  (void)hipMemsetAsync(relkT, 0, (size_t)HDK * LPAD * 2, stream);  // zero t-pad

  // rel_k = positions @ Wrel^T  -> relkT[f][t]
  gemm_wmma<GEMM_RELK><<<dim3(HDK / 16, (LREL + 63) / 64), 128, 0, stream>>>(
      pos, NRPF, WrT, HDK, LREL, NRPF, relkT, nullptr, nullptr, nullptr, nullptr, 1.f);
  // Q (scaled, two biased copies), K (transposed), V
  gemm_wmma<GEMM_Q><<<dim3(HDK / 16, M / 64), 128, 0, stream>>>(
      xh, DIM, WqT, HDK, M, DIM, qcb, qpb, nullptr, rcb, rpb, 0.125f);
  gemm_wmma<GEMM_K><<<dim3(HDK / 16, M / 64), 128, 0, stream>>>(
      xh, DIM, WkT, HDK, M, DIM, khT, nullptr, nullptr, nullptr, nullptr, 1.f);
  gemm_wmma<GEMM_V><<<dim3(HDV / 16, M / 64), 128, 0, stream>>>(
      xh, DIM, WvT, HDV, M, DIM, vhb, nullptr, nullptr, nullptr, nullptr, 1.f);
  // fused relative attention
  attn_kernel<<<dim3(SEQ / 16, BH), 128, 0, stream>>>(qcb, qpb, khT, vhb, relkT, outh);
  // output projection + bias (f32 out)
  gemm_wmma<GEMM_OUT><<<dim3(DIM / 16, M / 64), 128, 0, stream>>>(
      outh, HDV, WoT, DIM, M, HDV, nullptr, nullptr, out, bout, nullptr, 1.f);
}
